// Seq2SeqEncoder_26439818674411
// MI455X (gfx1250) — matrix operand order repaired and checked by reading, not compiled, not run on hardware
//
#include <hip/hip_runtime.h>
#include <hip/hip_bf16.h>

// ---------------- constants (match reference) ----------------
#define BB 128
#define TT 256
#define EE 512
#define HH 512
#define GG 2048   // 4*H
#define LL 4

typedef __attribute__((ext_vector_type(16))) __bf16 bf16x16;
typedef __attribute__((ext_vector_type(8)))  float  f32x8;

union Frag16 { bf16x16 v; uint4 q[2]; };

__device__ __forceinline__ unsigned short f2bf(float f) {
    union { float f; unsigned u; } v; v.f = f;
    unsigned r = v.u + 0x7FFFu + ((v.u >> 16) & 1u);   // round-to-nearest-even
    return (unsigned short)(r >> 16);
}
__device__ __forceinline__ float bf2f(unsigned short h) {
    union { unsigned u; float f; } v; v.u = ((unsigned)h) << 16; return v.f;
}

// A-matrix fragment (16x32 bf16): lane m=lane&15 holds row m; K-halves by lane>>4.
// VGPR0..3 <- K = k0+kh*8 .. +7 ; VGPR4..7 <- K = k0+16+kh*8 .. +7  (ISA 7.12.2)
__device__ __forceinline__ bf16x16 loadA(const unsigned short* base, int ld,
                                         int m, int k0, int kh) {
    Frag16 f;
    const unsigned short* p = base + (size_t)m * ld + k0 + kh * 8;
    f.q[0] = *(const uint4*)p;
    f.q[1] = *(const uint4*)(p + 16);
    return f.v;
}

// B-matrix fragment (32x16 bf16): lane n=lane&15 holds column n; K = k0+kh*8 .. +7 and
// k0+16+kh*8 .. +7 (same as A). W stored row-major [N][K] so K values are contiguous.
__device__ __forceinline__ bf16x16 loadB(const unsigned short* base, int ld,
                                         int n_glob, int k0, int kh) {
    Frag16 f;
    const unsigned short* p = base + (size_t)n_glob * ld + k0 + kh * 8;
    f.q[0] = *(const uint4*)p;
    f.q[1] = *(const uint4*)(p + 16);
    return f.v;
}

// ---------------- kernel 1: embedding gather + relu + bf16 ----------------
// x[t*B + b][e], one block per (t,b)
__global__ void embed_relu_kernel(const int* __restrict__ enc,
                                  const float* __restrict__ emb,
                                  unsigned short* __restrict__ x) {
    const int t = blockIdx.x / BB;
    const int b = blockIdx.x % BB;
    const int idx = enc[b * TT + t];
    const float* row = emb + (size_t)idx * EE;
    unsigned short* dst = x + (size_t)blockIdx.x * EE;
    for (int e = threadIdx.x; e < EE; e += blockDim.x) {
        float v = row[e];
        dst[e] = f2bf(v > 0.f ? v : 0.f);
    }
}

// ---------------- kernel 2: fp32 -> bf16 convert ----------------
__global__ void cvt_bf16_kernel(const float* __restrict__ s,
                                unsigned short* __restrict__ d, int n) {
    int i = blockIdx.x * blockDim.x + threadIdx.x;
    if (i < n) d[i] = f2bf(s[i]);
}

__global__ void bias_sum_kernel(const float* __restrict__ a,
                                const float* __restrict__ b,
                                float* __restrict__ d, int n) {
    int i = blockIdx.x * blockDim.x + threadIdx.x;
    if (i < n) d[i] = a[i] + b[i];
}

// ---------------- kernel 3: fused bidirectional residual LSTM ----------------
// grid = (8 batch tiles, 2 directions), block = 256 threads = 8 waves.
// Dynamic LDS: inp[16][512] bf16 | h[L][16][512] bf16 | c[L][16][512] f32  (208 KB)
__global__ void lstm_kernel(const unsigned short* __restrict__ x,     // [T][B][E]
                            const unsigned short* __restrict__ wih,   // [L][G][E] bf16
                            const unsigned short* __restrict__ whh,   // [L][G][H] bf16
                            const float* __restrict__ bsum,           // [L][G]
                            unsigned short* __restrict__ outcat,      // [B][T][2H] bf16
                            unsigned short* __restrict__ hTcat,       // [L][B][2H] bf16
                            unsigned short* __restrict__ cTcat) {     // [L][B][2H] bf16
    extern __shared__ char smem[];
    unsigned short* inp = (unsigned short*)smem;                 // 16*512
    unsigned short* hst = inp + 16 * 512;                        // L*16*512
    float*          cst = (float*)(hst + LL * 16 * 512);         // L*16*512

    const int tid  = threadIdx.x;
    const int wv   = tid >> 5;
    const int lane = tid & 31;
    const int m16  = lane & 15;
    const int kh   = lane >> 4;
    const int b0   = blockIdx.x * 16;
    const int dir  = blockIdx.y;
    const int j0w  = wv * 64;   // this wave owns hidden columns [j0w, j0w+64)

    for (int i = tid; i < LL * 16 * 512; i += 256) { hst[i] = 0; cst[i] = 0.f; }
    __syncthreads();

    for (int step = 0; step < TT; ++step) {
        const int t = dir ? (TT - 1 - step) : step;

        // stage x[t][b0..b0+16][*] into LDS (residual/input buffer)
        {
            const uint4* s4 = (const uint4*)(x + ((size_t)t * BB + b0) * EE);
            uint4* d4 = (uint4*)inp;
            for (int i = tid; i < 16 * 512 / 8; i += 256) d4[i] = s4[i];
        }
        __syncthreads();

        for (int l = 0; l < LL; ++l) {
            const unsigned short* wihl = wih + (size_t)l * GG * EE;
            const unsigned short* whhl = whh + (size_t)l * GG * HH;
            const unsigned short* hl   = hst + l * 16 * 512;
            unsigned short*       hlw  = hst + l * 16 * 512;
            float*                cl   = cst + l * 16 * 512;
            const float*          bl   = bsum + l * GG;

            // acc[g4*4+jt]: gate-group g4 (i,f,g,o), column tile jt of this wave
            f32x8 acc[16];
            const f32x8 zf = {0.f, 0.f, 0.f, 0.f, 0.f, 0.f, 0.f, 0.f};
#pragma unroll
            for (int i = 0; i < 16; ++i) acc[i] = zf;

            for (int k0 = 0; k0 < 512; k0 += 32) {
                bf16x16 aI = loadA(inp, 512, m16, k0, kh);
                bf16x16 aH = loadA(hl,  512, m16, k0, kh);
                if (k0 + 32 < 512)
                    __builtin_prefetch(wihl + (size_t)j0w * 512 + k0 + 32, 0, 0);
#pragma unroll
                for (int g4 = 0; g4 < 4; ++g4) {
#pragma unroll
                    for (int jt = 0; jt < 4; ++jt) {
                        const int n0 = g4 * 512 + j0w + jt * 16;
                        bf16x16 b1 = loadB(wihl, 512, n0 + m16, k0, kh);
                        acc[g4 * 4 + jt] = __builtin_amdgcn_wmma_f32_16x16x32_bf16(
                            false, aI, false, b1, (short)0, acc[g4 * 4 + jt], false, false);
                        bf16x16 b2 = loadB(whhl, 512, n0 + m16, k0, kh);
                        acc[g4 * 4 + jt] = __builtin_amdgcn_wmma_f32_16x16x32_bf16(
                            false, aH, false, b2, (short)0, acc[g4 * 4 + jt], false, false);
                    }
                }
            }
            __syncthreads();   // all waves done reading inp / h[l]

            // LSTM cell update: i/f/g/o for column j are all in this wave
#pragma unroll
            for (int jt = 0; jt < 4; ++jt) {
                const int j  = j0w + jt * 16 + m16;
                const float bi = bl[0 * 512 + j];
                const float bf = bl[1 * 512 + j];
                const float bg = bl[2 * 512 + j];
                const float bo = bl[3 * 512 + j];
                f32x8 vi = acc[0 * 4 + jt], vf = acc[1 * 4 + jt];
                f32x8 vg = acc[2 * 4 + jt], vo = acc[3 * 4 + jt];
#pragma unroll
                for (int r = 0; r < 8; ++r) {
                    const int m = r + kh * 8;             // C/D layout: lanes16-31 -> M+8
                    const int o = m * 512 + j;
                    float si = 1.f / (1.f + __expf(-(vi[r] + bi)));
                    float sf = 1.f / (1.f + __expf(-(vf[r] + bf)));
                    float so = 1.f / (1.f + __expf(-(vo[r] + bo)));
                    float tg = tanhf(vg[r] + bg);
                    float cn = sf * cl[o] + si * tg;
                    float hn = so * tanhf(cn);
                    cl[o]  = cn;
                    hlw[o] = f2bf(hn);
                    inp[o] = f2bf(hn + bf2f(inp[o]));     // residual: inp = h + inp
                }
            }
            __syncthreads();   // updated inp/h visible before next layer's GEMM
        }

        // write direction output: outcat[b][t][dir*H + j]
        {
            unsigned short* dst = outcat + (((size_t)b0) * TT + t) * (2 * HH) + dir * HH;
            for (int i = tid; i < 16 * 512 / 8; i += 256) {
                int row = i >> 6, col = i & 63;
                ((uint4*)(dst + (size_t)row * TT * 2 * HH))[col] =
                    ((const uint4*)(inp + row * 512))[col];
            }
        }
        __syncthreads();
    }

    // final h/c states -> hTcat/cTcat [L][B][2H]
    for (int i = tid; i < LL * 16 * 512; i += 256) {
        int l = i / (16 * 512), rem = i % (16 * 512);
        int m = rem / 512, j = rem % 512;
        size_t o = (((size_t)l * BB) + b0 + m) * (2 * HH) + (size_t)dir * HH + j;
        hTcat[o] = hst[i];
        cTcat[o] = f2bf(cst[i]);
    }
}

// ---------------- kernel 4: WMMA projection  C[M][512] = A[M][1024] @ W[512][1024]^T + b
__global__ void proj_kernel(const unsigned short* __restrict__ A,
                            const unsigned short* __restrict__ W,
                            const float* __restrict__ bias,
                            float* __restrict__ C, int M) {
    const int tid  = threadIdx.x;
    const int lane = tid & 31;
    const int m16  = lane & 15;
    const int kh   = lane >> 4;
    const int gw   = blockIdx.x * 8 + (tid >> 5);
    const int nt   = gw & 31;          // 512/16 = 32 column tiles
    const int mt   = gw >> 5;
    if (mt * 16 >= M) return;
    const int m0 = mt * 16, n0 = nt * 16;

    f32x8 acc = {0.f, 0.f, 0.f, 0.f, 0.f, 0.f, 0.f, 0.f};
    for (int k0 = 0; k0 < 1024; k0 += 32) {
        bf16x16 a = loadA(A + (size_t)m0 * 1024, 1024, m16, k0, kh);
        bf16x16 b = loadB(W, 1024, n0 + m16, k0, kh);
        acc = __builtin_amdgcn_wmma_f32_16x16x32_bf16(false, a, false, b,
                                                      (short)0, acc, false, false);
    }
    const float bn = bias[n0 + m16];
#pragma unroll
    for (int r = 0; r < 8; ++r) {
        int m = m0 + r + kh * 8;
        C[(size_t)m * 512 + n0 + m16] = acc[r] + bn;
    }
}

// ---------------- launch ----------------
extern "C" void kernel_launch(void* const* d_in, const int* in_sizes, int n_in,
                              void* d_out, int out_size, void* d_ws, size_t ws_size,
                              hipStream_t stream) {
    const int*   enc    = (const int*)d_in[0];
    const float* emb    = (const float*)d_in[1];
    const float* W_ih   = (const float*)d_in[2];
    const float* W_hh   = (const float*)d_in[3];
    const float* b_ih   = (const float*)d_in[4];
    const float* b_hh   = (const float*)d_in[5];
    const float* out_W  = (const float*)d_in[6];
    const float* out_b  = (const float*)d_in[7];
    const float* hid_W  = (const float*)d_in[8];
    const float* hid_b  = (const float*)d_in[9];
    const float* cell_W = (const float*)d_in[10];
    const float* cell_b = (const float*)d_in[11];
    float* out = (float*)d_out;

    // workspace partition (bytes, 256-aligned)
    char* ws = (char*)d_ws;
    size_t off = 0;
    auto alloc = [&](size_t bytes) { char* p = ws + off; off = (off + bytes + 255) & ~(size_t)255; return p; };
    unsigned short* x_bf    = (unsigned short*)alloc((size_t)TT * BB * EE * 2);      // 33.5 MB
    unsigned short* wih_bf  = (unsigned short*)alloc((size_t)LL * GG * EE * 2);      // 8.4 MB
    unsigned short* whh_bf  = (unsigned short*)alloc((size_t)LL * GG * HH * 2);      // 8.4 MB
    unsigned short* outW_bf = (unsigned short*)alloc((size_t)512 * 1024 * 2);
    unsigned short* hidW_bf = (unsigned short*)alloc((size_t)512 * 1024 * 2);
    unsigned short* celW_bf = (unsigned short*)alloc((size_t)512 * 1024 * 2);
    float*          bsum    = (float*)alloc((size_t)LL * GG * 4);
    unsigned short* outcat  = (unsigned short*)alloc((size_t)BB * TT * 2 * HH * 2);  // 67 MB
    unsigned short* hTcat   = (unsigned short*)alloc((size_t)LL * BB * 2 * HH * 2);
    unsigned short* cTcat   = (unsigned short*)alloc((size_t)LL * BB * 2 * HH * 2);
    (void)ws_size; (void)in_sizes; (void)n_in; (void)out_size;

    // 1) embedding + relu + bf16
    embed_relu_kernel<<<TT * BB, 256, 0, stream>>>(enc, emb, x_bf);

    // 2) weight conversions
    {
        int n = LL * GG * EE;
        cvt_bf16_kernel<<<(n + 255) / 256, 256, 0, stream>>>(W_ih, wih_bf, n);
        cvt_bf16_kernel<<<(n + 255) / 256, 256, 0, stream>>>(W_hh, whh_bf, n);
        int m = 512 * 1024;
        cvt_bf16_kernel<<<(m + 255) / 256, 256, 0, stream>>>(out_W,  outW_bf, m);
        cvt_bf16_kernel<<<(m + 255) / 256, 256, 0, stream>>>(hid_W,  hidW_bf, m);
        cvt_bf16_kernel<<<(m + 255) / 256, 256, 0, stream>>>(cell_W, celW_bf, m);
        int g = LL * GG;
        bias_sum_kernel<<<(g + 255) / 256, 256, 0, stream>>>(b_ih, b_hh, bsum, g);
    }

    // 3) fused bidirectional LSTM: 8 batch tiles x 2 directions, 208 KB LDS each
    {
        size_t lds = (size_t)16 * 512 * 2 + (size_t)LL * 16 * 512 * 2 + (size_t)LL * 16 * 512 * 4;
        lstm_kernel<<<dim3(BB / 16, 2), 256, lds, stream>>>(x_bf, wih_bf, whh_bf, bsum,
                                                            outcat, hTcat, cTcat);
    }

    // 4) projections
    {
        int M = BB * TT;                                  // out: 32768 x 512
        proj_kernel<<<(M / 16) * 32 / 8, 256, 0, stream>>>(outcat, outW_bf, out_b, out, M);
        float* hid_out  = out + (size_t)M * 512;          // hidden: 512 x 512
        float* cell_out = hid_out + (size_t)LL * BB * 512;
        proj_kernel<<<(LL * BB / 16) * 32 / 8, 256, 0, stream>>>(hTcat, hidW_bf, hid_b, hid_out,  LL * BB);
        proj_kernel<<<(LL * BB / 16) * 32 / 8, 256, 0, stream>>>(cTcat, celW_bf, cell_b, cell_out, LL * BB);
    }
}
